// TransformerBlock_28286654612086
// MI455X (gfx1250) — matrix operand order repaired and checked by reading, not compiled, not run on hardware
//
#include <hip/hip_runtime.h>

// ---------------- problem constants ----------------
#define BB 4
#define NN 4096
#define DD 256
#define KK 16
#define BN_ROWS (BB * NN)          // 16384
#define PE_IN 33

// ---------------- WMMA vector types ----------------
typedef __attribute__((ext_vector_type(16))) __bf16 v16bf;
typedef __attribute__((ext_vector_type(8)))  __bf16 v8bf;
typedef __attribute__((ext_vector_type(8)))  float  v8f;

// LDS pitches (bf16 elems / f32 elems) chosen so 16 consecutive rows hit
// distinct bank groups for b128 reads.
#define APITCH 264   // bf16 row pitch (256 + 8)
#define VPITCH 260   // f32  row pitch (256 + 4)
#define PEP    36    // pos-embed row pitch (33 padded)

__device__ __forceinline__ __bf16 to_bf16(float x) {
    unsigned u = __builtin_bit_cast(unsigned, x);
    unsigned r = (u + 0x7fffu + ((u >> 16) & 1u)) >> 16;   // round-nearest-even
    unsigned short h = (unsigned short)r;
    return __builtin_bit_cast(__bf16, h);
}

// A fragment (16x32 bf16) from LDS row-major [rows][APITCH].
// lane m=l&15; lanes<16 -> K {k0..k0+15},
// lanes>=16 -> K {k0+16..k0+31} (matches B fragment order).
__device__ __forceinline__ v16bf load_a_frag(const __bf16* s, int m0, int k0, int lane) {
    int m = m0 + (lane & 15);
    int kb = (lane >> 4) << 4;              // 0 or 16
    const __bf16* p = s + m * APITCH + k0 + kb;
    v8bf lo = *(const v8bf*)p;
    v8bf hi = *(const v8bf*)(p + 8);
    v16bf a;
#pragma unroll
    for (int i = 0; i < 8; ++i) { a[i] = lo[i]; a[8 + i] = hi[i]; }
    return a;
}

// B fragment (32x16 bf16) for D = A @ W^T, W row-major [256][256] bf16.
// b[k][n] = W[n0+n][k0+k]; lane n=l&15, K-half by l>=16 -> 32 contiguous bytes.
__device__ __forceinline__ v16bf load_b_frag(const __bf16* W, int n0, int k0, int lane) {
    int n  = lane & 15;
    int kb = (lane >> 4) << 4;              // 0 or 16
    return *(const v16bf*)(W + (size_t)(n0 + n) * DD + k0 + kb);
}

// Two N-tiles per pass, double-buffered B: loads for step k+1 are issued
// before the WMMAs of step k, and each A fragment feeds two independent
// WMMA chains so load latency overlaps the XDL pipe.
__device__ __forceinline__ void gemm_pair(const __bf16* sA, int m0,
                                          const __bf16* W, int nt0, int lane,
                                          v8f& acc0, v8f& acc1) {
    acc0 = {}; acc1 = {};
    v16bf b0 = load_b_frag(W, (nt0 + 0) * 16, 0, lane);
    v16bf b1 = load_b_frag(W, (nt0 + 1) * 16, 0, lane);
#pragma unroll
    for (int kk = 0; kk < 8; ++kk) {
        v16bf nb0, nb1;
        if (kk < 7) {
            nb0 = load_b_frag(W, (nt0 + 0) * 16, (kk + 1) * 32, lane);
            nb1 = load_b_frag(W, (nt0 + 1) * 16, (kk + 1) * 32, lane);
        }
        v16bf a = load_a_frag(sA, m0, kk * 32, lane);
        acc0 = __builtin_amdgcn_wmma_f32_16x16x32_bf16(false, a, false, b0,
                                                       (short)0, acc0, false, false);
        acc1 = __builtin_amdgcn_wmma_f32_16x16x32_bf16(false, a, false, b1,
                                                       (short)0, acc1, false, false);
        if (kk < 7) { b0 = nb0; b1 = nb1; }
    }
}

// ---------------- f32 -> bf16 convert ----------------
__global__ void cvt_bf16(const float* __restrict__ s, __bf16* __restrict__ d, int n) {
    int i = blockIdx.x * 256 + threadIdx.x;
    if (i < n) d[i] = to_bf16(s[i]);
}

// ---------------- KNN (top-16 by squared distance) ----------------
__global__ void knn_kernel(const float* __restrict__ xyz, int* __restrict__ knn) {
    __shared__ float sx[NN], sy[NN], sz[NN];
    int b  = blockIdx.x >> 5;               // 32 blocks per batch
    int q0 = (blockIdx.x & 31) * 128;
    const float* base = xyz + (size_t)b * NN * 3;
    for (int i = threadIdx.x; i < NN; i += 128) {
        sx[i] = base[i * 3 + 0]; sy[i] = base[i * 3 + 1]; sz[i] = base[i * 3 + 2];
    }
    __syncthreads();
    int qi = q0 + threadIdx.x;
    float qx = sx[qi], qy = sy[qi], qz = sz[qi];
    float bd[KK]; int bi[KK];
#pragma unroll
    for (int i = 0; i < KK; ++i) { bd[i] = 3.4e38f; bi[i] = 0; }
    for (int j = 0; j < NN; ++j) {
        float dx = qx - sx[j], dy = qy - sy[j], dz = qz - sz[j];
        float d2 = dx * dx + dy * dy + dz * dz;
        if (d2 < bd[KK - 1]) {
            int pos = KK - 1;
            while (pos > 0 && bd[pos - 1] > d2) {
                bd[pos] = bd[pos - 1]; bi[pos] = bi[pos - 1]; --pos;
            }
            bd[pos] = d2; bi[pos] = j;
        }
    }
    int* o = knn + ((size_t)b * NN + qi) * KK;
#pragma unroll
    for (int i = 0; i < KK; ++i) o[i] = bi[i];
}

// ---------------- Q/K/V projection: WMMA bf16 GEMMs ----------------
__global__ void __launch_bounds__(256, 1)
qkv_kernel(const __bf16* __restrict__ fb,
           const __bf16* __restrict__ wq, const __bf16* __restrict__ wk,
           const __bf16* __restrict__ wv,
           float* __restrict__ q, float* __restrict__ k, float* __restrict__ v) {
    __shared__ __bf16 sA[64 * APITCH];
    int gm0 = blockIdx.x * 64;
    for (int i = threadIdx.x; i < 64 * DD; i += 256) {
        int r = i >> 8, c = i & 255;
        sA[r * APITCH + c] = fb[(size_t)(gm0 + r) * DD + c];
    }
    __syncthreads();
    int wave = threadIdx.x >> 5, lane = threadIdx.x & 31;
    int mt = wave & 3, half = wave >> 2;
    int n = lane & 15, mh = (lane >> 4) << 3;
#pragma unroll
    for (int wsel = 0; wsel < 3; ++wsel) {     // unrolled: W/O are compile-time per copy
        const __bf16* W = (wsel == 0) ? wq : (wsel == 1 ? wk : wv);
        float* O        = (wsel == 0) ? q  : (wsel == 1 ? k  : v);
        for (int ntp = 0; ntp < 4; ++ntp) {
            int nt0 = half * 8 + ntp * 2;
            v8f acc0, acc1;
            gemm_pair(sA, mt * 16, W, nt0, lane, acc0, acc1);
#pragma unroll
            for (int r = 0; r < 8; ++r) {
                size_t row = (size_t)(gm0 + mt * 16 + mh + r) * DD;
                O[row + (nt0 + 0) * 16 + n] = acc0[r];
                O[row + (nt0 + 1) * 16 + n] = acc1[r];
            }
        }
    }
}

// ---------------- fused edge-MLP / softmax / aggregate ----------------
// One block = G=4 points = 64 edge rows = 4 WMMA M-tiles, 8 waves.
#define G 4
#define ROWS 64
__global__ void __launch_bounds__(256, 1)
fused_kernel(const float* __restrict__ xyz, const float* __restrict__ feats,
             const float* __restrict__ qb, const float* __restrict__ kb,
             const float* __restrict__ vb, const int* __restrict__ knn,
             const __bf16* __restrict__ gw1, const __bf16* __restrict__ gw2,
             const float* __restrict__ gb1, const float* __restrict__ gb2,
             const float* __restrict__ pew, const float* __restrict__ peb,
             float* __restrict__ res) {
    extern __shared__ char smem[];
    __bf16* sE  = (__bf16*)smem;                          // edges bf16 [ROWS][APITCH]
    __bf16* sH  = sE + ROWS * APITCH;                     // hidden bf16
    float*  sV  = (float*)(sH + ROWS * APITCH);           // v + pos_encode f32 [ROWS][VPITCH]
    float*  sPE = sV + ROWS * VPITCH;                     // pos_embed [ROWS][PEP]
    int*    sIdx = (int*)(sPE + ROWS * PEP);              // [ROWS]

    int tid = threadIdx.x;
    int gp0 = blockIdx.x * G;                             // global row (b*N+n), batch-aligned

    // ---- phase 1: indices + Fourier positional embedding (rows mapped to t<64)
    if (tid < ROWS) {
        int r = tid, p = r >> 4, s = r & 15;
        int gp = gp0 + p, b = gp >> 12, n = gp & (NN - 1);
        int j = knn[((size_t)b * NN + n) * KK + s];
        sIdx[r] = j;
        const float* pc = xyz + ((size_t)b * NN + n) * 3;
        const float* nc = xyz + ((size_t)b * NN + j) * 3;
#pragma unroll
        for (int d = 0; d < 3; ++d) {
            float diff = pc[d] - nc[d];
            sPE[r * PEP + d] = diff;
#pragma unroll
            for (int f = 0; f < 5; ++f) {                  // linspace(1, 32, 5)
                float fr = 1.0f + 7.75f * (float)f;
                sPE[r * PEP + 3 + 6 * f + d]     = __sinf(diff * fr);
                sPE[r * PEP + 3 + 6 * f + 3 + d] = __cosf(diff * fr);
            }
        }
    }
    __syncthreads();

    // ---- phase 2: pos_encode + gathers -> edges(bf16), v+pos(f32). thread = channel c.
    {
        int c = tid;
        float pw[PE_IN];
#pragma unroll
        for (int i = 0; i < PE_IN; ++i) pw[i] = pew[c * PE_IN + i];
        float pb = peb[c];
#pragma unroll
        for (int p = 0; p < G; ++p) {
            int gp = gp0 + p, b = gp >> 12;
            float qv = qb[(size_t)gp * DD + c];
            for (int s = 0; s < KK; ++s) {
                int r = p * KK + s;
                int j = sIdx[r];
                float pe = pb;
#pragma unroll
                for (int i = 0; i < PE_IN; ++i) pe += pw[i] * sPE[r * PEP + i];
                size_t gidx = ((size_t)b * NN + j) * DD + c;
                float kv = kb[gidx], vv = vb[gidx];
                sE[r * APITCH + c] = to_bf16(qv - kv + pe);
                sV[r * VPITCH + c] = vv + pe;
            }
        }
    }
    __syncthreads();

    int wave = tid >> 5, lane = tid & 31;
    int mt = wave & 3, half = wave >> 2;                   // wave -> (M-tile, N-half)
    int ncol = lane & 15, mh = (lane >> 4) << 3;

    // ---- phase 3: GEMM1 (edges @ g_w1^T) + bias + relu -> sH (bf16)
    for (int ntp = 0; ntp < 4; ++ntp) {
        int nt0 = half * 8 + ntp * 2;
        v8f acc0, acc1;
        gemm_pair(sE, mt * 16, gw1, nt0, lane, acc0, acc1);
#pragma unroll
        for (int t2 = 0; t2 < 2; ++t2) {
            int nt = nt0 + t2;
            float bias = gb1[nt * 16 + ncol];
#pragma unroll
            for (int r = 0; r < 8; ++r) {
                float hv = (t2 ? acc1[r] : acc0[r]) + bias;
                hv = hv > 0.0f ? hv : 0.0f;
                sH[(mt * 16 + mh + r) * APITCH + nt * 16 + ncol] = to_bf16(hv);
            }
        }
    }
    __syncthreads();

    // ---- phase 4: GEMM2 + in-register neighbor softmax + aggregate + residual
    int gp = gp0 + mt;
    for (int ntp = 0; ntp < 4; ++ntp) {
        int nt0 = half * 8 + ntp * 2;
        v8f acc0, acc1;
        gemm_pair(sH, mt * 16, gw2, nt0, lane, acc0, acc1);
#pragma unroll
        for (int t2 = 0; t2 < 2; ++t2) {
            int nt = nt0 + t2;
            float bias = gb2[nt * 16 + ncol];
            float av[8], mx = -3.4e38f;
#pragma unroll
            for (int r = 0; r < 8; ++r) {
                av[r] = (t2 ? acc1[r] : acc0[r]) + bias;
                mx = fmaxf(mx, av[r]);
            }
            mx = fmaxf(mx, __shfl_xor(mx, 16));            // join the two 8-row halves
            float ssum = 0.0f;
#pragma unroll
            for (int r = 0; r < 8; ++r) { av[r] = __expf(av[r] - mx); ssum += av[r]; }
            ssum += __shfl_xor(ssum, 16);
            float inv = 1.0f / ssum;
            float part = 0.0f;
#pragma unroll
            for (int r = 0; r < 8; ++r)
                part += av[r] * inv * sV[(mt * 16 + mh + r) * VPITCH + nt * 16 + ncol];
            part += __shfl_xor(part, 16);
            if (lane < 16) {
                int f = nt * 16 + ncol;
                res[(size_t)gp * DD + f] = part + feats[(size_t)gp * DD + f];
            }
        }
    }
}

// ---------------- deterministic BatchNorm ----------------
__global__ void bn_part(const float* __restrict__ res, float* __restrict__ psum,
                        float* __restrict__ psq) {
    int c = threadIdx.x, r0 = blockIdx.x * 64;
    float s = 0.0f, s2 = 0.0f;
    for (int i = 0; i < 64; ++i) {
        float v = res[(size_t)(r0 + i) * DD + c];
        s += v; s2 += v * v;
    }
    psum[blockIdx.x * DD + c] = s;
    psq [blockIdx.x * DD + c] = s2;
}

__global__ void bn_final(const float* __restrict__ psum, const float* __restrict__ psq,
                         const float* __restrict__ bnw, const float* __restrict__ bnb,
                         float* __restrict__ stats) {
    int c = threadIdx.x;
    float s = 0.0f, s2 = 0.0f;
    for (int i = 0; i < 256; ++i) { s += psum[i * DD + c]; s2 += psq[i * DD + c]; }
    float mean = s * (1.0f / (float)BN_ROWS);
    float var  = s2 * (1.0f / (float)BN_ROWS) - mean * mean;   // biased, as jnp.var
    float sc = bnw[c] * rsqrtf(var + 1e-5f);
    stats[c] = sc;
    stats[DD + c] = bnb[c] - mean * sc;
}

__global__ void bn_apply(const float* __restrict__ res, const float* __restrict__ stats,
                         float* __restrict__ out) {
    int i = blockIdx.x * 256 + threadIdx.x;
    int c = i & (DD - 1);
    out[i] = res[i] * stats[c] + stats[DD + c];
}

// ---------------- host launch ----------------
extern "C" void kernel_launch(void* const* d_in, const int* in_sizes, int n_in,
                              void* d_out, int out_size, void* d_ws, size_t ws_size,
                              hipStream_t stream) {
    const float* xyz   = (const float*)d_in[0];
    const float* feats = (const float*)d_in[1];
    const float* w_q   = (const float*)d_in[2];
    const float* w_k   = (const float*)d_in[3];
    const float* w_v   = (const float*)d_in[4];
    const float* g_w1  = (const float*)d_in[5];
    const float* g_b1  = (const float*)d_in[6];
    const float* g_w2  = (const float*)d_in[7];
    const float* g_b2  = (const float*)d_in[8];
    const float* pe_w  = (const float*)d_in[9];
    const float* pe_b  = (const float*)d_in[10];
    const float* bn_w  = (const float*)d_in[11];
    const float* bn_b  = (const float*)d_in[12];
    float* out = (float*)d_out;

    char* ws = (char*)d_ws;
    size_t off = 0;
    auto alloc = [&](size_t bytes) -> void* {
        off = (off + 255) & ~(size_t)255;
        void* p = ws + off;
        off += bytes;
        return p;
    };
    const size_t nFeat = (size_t)BN_ROWS * DD;        // 4,194,304
    int*    knn  = (int*)   alloc((size_t)BN_ROWS * KK * 4);
    float*  qb   = (float*) alloc(nFeat * 4);
    float*  kb   = (float*) alloc(nFeat * 4);
    float*  vb   = (float*) alloc(nFeat * 4);
    float*  resb = (float*) alloc(nFeat * 4);
    __bf16* fb   = (__bf16*)alloc(nFeat * 2);
    __bf16* wqb  = (__bf16*)alloc((size_t)DD * DD * 2);
    __bf16* wkb  = (__bf16*)alloc((size_t)DD * DD * 2);
    __bf16* wvb  = (__bf16*)alloc((size_t)DD * DD * 2);
    __bf16* g1b  = (__bf16*)alloc((size_t)DD * DD * 2);
    __bf16* g2b  = (__bf16*)alloc((size_t)DD * DD * 2);
    float*  psum = (float*) alloc(256 * DD * 4);
    float*  psq  = (float*) alloc(256 * DD * 4);
    float*  stats= (float*) alloc(2 * DD * 4);
    (void)ws_size; (void)n_in; (void)in_sizes; (void)out_size;

    // bf16 conversions
    cvt_bf16<<<(int)((nFeat + 255) / 256), 256, 0, stream>>>(feats, fb, (int)nFeat);
    cvt_bf16<<<(DD * DD + 255) / 256, 256, 0, stream>>>(w_q,  wqb, DD * DD);
    cvt_bf16<<<(DD * DD + 255) / 256, 256, 0, stream>>>(w_k,  wkb, DD * DD);
    cvt_bf16<<<(DD * DD + 255) / 256, 256, 0, stream>>>(w_v,  wvb, DD * DD);
    cvt_bf16<<<(DD * DD + 255) / 256, 256, 0, stream>>>(g_w1, g1b, DD * DD);
    cvt_bf16<<<(DD * DD + 255) / 256, 256, 0, stream>>>(g_w2, g2b, DD * DD);

    knn_kernel<<<BB * (NN / 128), 128, 0, stream>>>(xyz, knn);
    qkv_kernel<<<BN_ROWS / 64, 256, 0, stream>>>(fb, wqb, wkb, wvb, qb, kb, vb);

    const size_t fusedSmem =
        (size_t)ROWS * APITCH * 2 * 2 +     // sE + sH (bf16)
        (size_t)ROWS * VPITCH * 4 +         // sV
        (size_t)ROWS * PEP * 4 +            // sPE
        (size_t)ROWS * 4;                   // sIdx   (~143.6 KB, fits 320 KB/WGP)
    fused_kernel<<<BN_ROWS / G, 256, fusedSmem, stream>>>(
        xyz, feats, qb, kb, vb, knn, g1b, g2b, g_b1, g_b2, pe_w, pe_b, resb);

    bn_part<<<256, 256, 0, stream>>>(resb, psum, psq);
    bn_final<<<1, 256, 0, stream>>>(psum, psq, bn_w, bn_b, stats);
    bn_apply<<<BN_ROWS, 256, 0, stream>>>(resb, stats, out);
}
